// LSTMAutoEncoder_30219389894832
// MI455X (gfx1250) — matrix operand order repaired and checked by reading, not compiled, not run on hardware
//
#include <hip/hip_runtime.h>

// Problem constants (match reference)
#define BDIM 256
#define TDIM 512
#define DDIM 32
#define HDIM 256

typedef __attribute__((ext_vector_type(16))) __bf16 v16bf;
typedef __attribute__((ext_vector_type(8)))  float  v8f;
typedef __attribute__((ext_vector_type(4)))  unsigned int v4u;

union Frag {
    v16bf bf;
    v4u   u[2];
};

__device__ __forceinline__ unsigned short f2bf(float x) {
    unsigned int u = __float_as_uint(x);
    unsigned int r = u + 0x7fffu + ((u >> 16) & 1u);   // round-to-nearest-even
    return (unsigned short)(r >> 16);
}
__device__ __forceinline__ float sigm(float x) { return 1.0f / (1.0f + __expf(-x)); }
__device__ __forceinline__ float tanh_fast(float x) { return 2.0f / (1.0f + __expf(-2.0f * x)) - 1.0f; }

// A fragment: 16x32 bf16, row-major source with leading dim `ld`.
// Lane L<16: M=L, K = k0+0..7 (v0..v3) and k0+16..23 (v4..v7)
// Lane L>=16: M=L-16, K = k0+8..15 and k0+24..31
__device__ __forceinline__ v16bf load_a(const unsigned short* __restrict__ base,
                                        int ld, int m0, int k0, int lane) {
    const int m  = m0 + (lane & 15);
    const int kb = k0 + ((lane >> 4) << 3);
    const unsigned short* p = base + (long)m * ld + kb;
    Frag f;
    f.u[0] = *(const v4u*)(p);        // K kb..kb+7
    f.u[1] = *(const v4u*)(p + 16);   // K kb+16..kb+23
    return f.bf;
}

// B fragment: 32x16 bf16 where B[k][n] = W[n0+n][k] (W row-major, leading dim ld).
// Lane L<16: N=L, K = k0+0..7 and k0+16..23 ; Lane L>=16: N=L-16, K = k0+8..15 and k0+24..31
__device__ __forceinline__ v16bf load_b(const unsigned short* __restrict__ W,
                                        int ld, int n0, int k0, int lane) {
    const int n  = n0 + (lane & 15);
    const int kb = k0 + ((lane >> 4) << 3);
    const unsigned short* p = W + (long)n * ld + kb;
    Frag f;
    f.u[0] = *(const v4u*)(p);        // K kb..kb+7
    f.u[1] = *(const v4u*)(p + 16);   // K kb+16..kb+23
    return f.bf;
}

// Fused LSTM cell step:  g = x@Wih.T + h@Whh.T + b ; gate nonlinearities; update c (f32, in
// place) and h (bf16, double-buffered).  Grid (HDIM/128, BDIM/16), block 256 (8 waves).
// Each wave: one 16-col h tile -> 4 gate tiles (i,f,g,o) held together so the cell update is
// purely local.
__global__ __launch_bounds__(256) void lstm_cell_kernel(
    const unsigned short* __restrict__ x, int x_ld, int Kx,      // bf16 [BDIM, Kx] (ld x_ld)
    const unsigned short* __restrict__ Wih,                      // bf16 [4H, Kx]
    const unsigned short* __restrict__ Whh,                      // bf16 [4H, H]
    const float*          __restrict__ bias,                     // f32  [4H] (b_ih+b_hh)
    const unsigned short* __restrict__ h_in,                     // bf16 [BDIM, H]
    unsigned short*       __restrict__ h_out,                    // bf16 [BDIM, H]
    float*                __restrict__ c)                        // f32  [BDIM, H] in/out
{
    const int lane = threadIdx.x & 31;
    const int wave = threadIdx.x >> 5;
    const int row0 = blockIdx.y << 4;
    const int col0 = (blockIdx.x << 7) + (wave << 4);

    v8f acc[4];
    #pragma unroll
    for (int g = 0; g < 4; ++g) acc[g] = (v8f)(0.0f);

    // input contribution (Kx = 32 or 256)
    for (int k0 = 0; k0 < Kx; k0 += 32) {
        v16bf a = load_a(x, x_ld, row0, k0, lane);
        #pragma unroll
        for (int g = 0; g < 4; ++g) {
            v16bf b = load_b(Wih, Kx, g * HDIM + col0, k0, lane);
            acc[g] = __builtin_amdgcn_wmma_f32_16x16x32_bf16(false, a, false, b,
                                                             (short)0, acc[g], false, false);
        }
    }
    // recurrent contribution (K = 256)
    for (int k0 = 0; k0 < HDIM; k0 += 32) {
        v16bf a = load_a(h_in, HDIM, row0, k0, lane);
        #pragma unroll
        for (int g = 0; g < 4; ++g) {
            v16bf b = load_b(Whh, HDIM, g * HDIM + col0, k0, lane);
            acc[g] = __builtin_amdgcn_wmma_f32_16x16x32_bf16(false, a, false, b,
                                                             (short)0, acc[g], false, false);
        }
    }

    // Epilogue: C/D layout => lane holds (m = mrow+v, n = ncol) for v in 0..7
    const int ncol = col0 + (lane & 15);
    const float bi = bias[0 * HDIM + ncol];
    const float bf = bias[1 * HDIM + ncol];
    const float bg = bias[2 * HDIM + ncol];
    const float bo = bias[3 * HDIM + ncol];
    const int mrow = row0 + ((lane >> 4) << 3);

    #pragma unroll
    for (int v = 0; v < 8; ++v) {
        const int idx = (mrow + v) * HDIM + ncol;
        float ig = sigm(acc[0][v] + bi);
        float fg = sigm(acc[1][v] + bf);
        float gg = tanh_fast(acc[2][v] + bg);
        float og = sigm(acc[3][v] + bo);
        float cn = fg * c[idx] + ig * gg;
        c[idx] = cn;
        h_out[idx] = f2bf(og * tanh_fast(cn));
    }
}

// Decoder output projection: pred = h1 @ fcW.T + fcb.  Writes f32 into the (time-reversed)
// output slot and bf16 into the autoregressive feedback buffer.  Grid 16, block 64 (2 waves).
__global__ __launch_bounds__(64) void fc_kernel(
    const unsigned short* __restrict__ h1,     // bf16 [BDIM, H]
    const unsigned short* __restrict__ fcW,    // bf16 [D, H]
    const float*          __restrict__ fcb,    // f32  [D]
    float*                __restrict__ out_t,  // f32, pre-offset to time slot; row stride T*D
    unsigned short*       __restrict__ xdec)   // bf16 [BDIM, D]
{
    const int lane = threadIdx.x & 31;
    const int wave = threadIdx.x >> 5;
    const int row0 = blockIdx.x << 4;
    const int col0 = wave << 4;

    v8f acc = (v8f)(0.0f);
    for (int k0 = 0; k0 < HDIM; k0 += 32) {
        v16bf a = load_a(h1, HDIM, row0, k0, lane);
        v16bf b = load_b(fcW, HDIM, col0, k0, lane);
        acc = __builtin_amdgcn_wmma_f32_16x16x32_bf16(false, a, false, b,
                                                      (short)0, acc, false, false);
    }
    const int n = col0 + (lane & 15);
    const float bn = fcb[n];
    const int mrow = row0 + ((lane >> 4) << 3);
    #pragma unroll
    for (int v = 0; v < 8; ++v) {
        const int m = mrow + v;
        const float val = acc[v] + bn;
        out_t[m * (TDIM * DDIM) + n] = val;
        xdec[m * DDIM + n] = f2bf(val);
    }
}

__global__ void f32_to_bf16_kernel(const float* __restrict__ s,
                                   unsigned short* __restrict__ d, int n) {
    int i = blockIdx.x * blockDim.x + threadIdx.x;
    if (i < n) d[i] = f2bf(s[i]);
}

__global__ void zero_u32_kernel(unsigned int* __restrict__ p, int n) {
    int i = blockIdx.x * blockDim.x + threadIdx.x;
    if (i < n) p[i] = 0u;
}

extern "C" void kernel_launch(void* const* d_in, const int* in_sizes, int n_in,
                              void* d_out, int out_size, void* d_ws, size_t ws_size,
                              hipStream_t stream)
{
    (void)in_sizes; (void)n_in; (void)out_size; (void)ws_size;

    char* ws = (char*)d_ws;
    size_t off = 0;
    auto alloc = [&](size_t bytes) -> void* {
        off = (off + 255) & ~(size_t)255;
        void* p = ws + off;
        off += bytes;
        return p;
    };
    const size_t hb = sizeof(unsigned short);

    // bf16 staging buffers
    unsigned short* xin   = (unsigned short*)alloc((size_t)BDIM * TDIM * DDIM * hb);
    unsigned short* eWih0 = (unsigned short*)alloc((size_t)4 * HDIM * DDIM * hb);
    unsigned short* eWhh0 = (unsigned short*)alloc((size_t)4 * HDIM * HDIM * hb);
    unsigned short* eWih1 = (unsigned short*)alloc((size_t)4 * HDIM * HDIM * hb);
    unsigned short* eWhh1 = (unsigned short*)alloc((size_t)4 * HDIM * HDIM * hb);
    unsigned short* dWih0 = (unsigned short*)alloc((size_t)4 * HDIM * DDIM * hb);
    unsigned short* dWhh0 = (unsigned short*)alloc((size_t)4 * HDIM * HDIM * hb);
    unsigned short* dWih1 = (unsigned short*)alloc((size_t)4 * HDIM * HDIM * hb);
    unsigned short* dWhh1 = (unsigned short*)alloc((size_t)4 * HDIM * HDIM * hb);
    unsigned short* fcWb  = (unsigned short*)alloc((size_t)DDIM * HDIM * hb);
    // states: h double-buffered (bf16), c in-place (f32)
    unsigned short* h0b[2];
    unsigned short* h1b[2];
    h0b[0] = (unsigned short*)alloc((size_t)BDIM * HDIM * hb);
    h0b[1] = (unsigned short*)alloc((size_t)BDIM * HDIM * hb);
    h1b[0] = (unsigned short*)alloc((size_t)BDIM * HDIM * hb);
    h1b[1] = (unsigned short*)alloc((size_t)BDIM * HDIM * hb);
    float* c0 = (float*)alloc((size_t)BDIM * HDIM * sizeof(float));
    float* c1 = (float*)alloc((size_t)BDIM * HDIM * sizeof(float));
    unsigned short* xdec = (unsigned short*)alloc((size_t)BDIM * DDIM * hb);

    // ---- zero initial state every call (deterministic; harness doesn't re-poison) ----
    auto zero = [&](void* p, size_t bytes) {
        int n = (int)(bytes / 4);
        zero_u32_kernel<<<(n + 255) / 256, 256, 0, stream>>>((unsigned int*)p, n);
    };
    zero(h0b[0], (size_t)BDIM * HDIM * hb);
    zero(h0b[1], (size_t)BDIM * HDIM * hb);
    zero(h1b[0], (size_t)BDIM * HDIM * hb);
    zero(h1b[1], (size_t)BDIM * HDIM * hb);
    zero(c0, (size_t)BDIM * HDIM * sizeof(float));
    zero(c1, (size_t)BDIM * HDIM * sizeof(float));
    zero(xdec, (size_t)BDIM * DDIM * hb);

    // ---- f32 -> bf16 conversions ----
    auto conv = [&](const void* s, unsigned short* d, int n) {
        f32_to_bf16_kernel<<<(n + 255) / 256, 256, 0, stream>>>((const float*)s, d, n);
    };
    conv(d_in[0],  xin,   BDIM * TDIM * DDIM);
    conv(d_in[1],  eWih0, 4 * HDIM * DDIM);
    conv(d_in[2],  eWhh0, 4 * HDIM * HDIM);
    conv(d_in[4],  eWih1, 4 * HDIM * HDIM);
    conv(d_in[5],  eWhh1, 4 * HDIM * HDIM);
    conv(d_in[7],  dWih0, 4 * HDIM * DDIM);
    conv(d_in[8],  dWhh0, 4 * HDIM * HDIM);
    conv(d_in[10], dWih1, 4 * HDIM * HDIM);
    conv(d_in[11], dWhh1, 4 * HDIM * HDIM);
    conv(d_in[13], fcWb,  DDIM * HDIM);

    const float* b_e0 = (const float*)d_in[3];
    const float* b_e1 = (const float*)d_in[6];
    const float* b_d0 = (const float*)d_in[9];
    const float* b_d1 = (const float*)d_in[12];
    const float* fc_b = (const float*)d_in[14];
    float* out = (float*)d_out;

    const dim3 cell_grid(HDIM / 128, BDIM / 16);

    int p = 0;
    // ---- encoder: layers interleaved per time step (both are forward scans) ----
    for (int t = 0; t < TDIM; ++t) {
        lstm_cell_kernel<<<cell_grid, 256, 0, stream>>>(
            xin + (size_t)t * DDIM, TDIM * DDIM, DDIM,
            eWih0, eWhh0, b_e0, h0b[p], h0b[p ^ 1], c0);
        lstm_cell_kernel<<<cell_grid, 256, 0, stream>>>(
            h0b[p ^ 1], HDIM, HDIM,
            eWih1, eWhh1, b_e1, h1b[p], h1b[p ^ 1], c1);
        p ^= 1;
    }
    // ---- decoder: autoregressive; (h,c) continue from encoder finals ----
    for (int t = 0; t < TDIM; ++t) {
        lstm_cell_kernel<<<cell_grid, 256, 0, stream>>>(
            xdec, DDIM, DDIM,
            dWih0, dWhh0, b_d0, h0b[p], h0b[p ^ 1], c0);
        lstm_cell_kernel<<<cell_grid, 256, 0, stream>>>(
            h0b[p ^ 1], HDIM, HDIM,
            dWih1, dWhh1, b_d1, h1b[p], h1b[p ^ 1], c1);
        fc_kernel<<<BDIM / 16, 64, 0, stream>>>(
            h1b[p ^ 1], fcWb, fc_b,
            out + (size_t)(TDIM - 1 - t) * DDIM, xdec);   // time-reversed output slot
        p ^= 1;
    }
}
